// MRBF_84808424227154
// MI455X (gfx1250) — matrix operand order repaired and checked by reading, not compiled, not run on hardware
//
#include <hip/hip_runtime.h>
#include <hip/hip_bf16.h>
#include <float.h>

typedef __attribute__((ext_vector_type(16))) _Float16 v16h;
typedef __attribute__((ext_vector_type(8)))  _Float16 v8h;
typedef __attribute__((ext_vector_type(8)))  float    v8f;

#define IN_F   32
#define OUT_F  128
#define ROW_H  40   // halves per centres row in LDS (32 padded to 40 -> 80B stride, conflict-free b128)
#define LOG2E  1.44269504088896340736f

// One block = 256 threads = 8 wave32; each wave owns a 16x128 output slab.
// Centres staged to LDS pre-scaled by -2 (exact in f16); WMMA accumulator
// initialized to x^2[row] + c^2[col], so one v_wmma_f32_16x16x32_f16 per tile
// yields sqdist directly: D = x * (-2c) + (x^2 + c^2).
// Epilogue: v_med3 clamp / mul / v_exp_f32 / store. NO bounds checks: the
// grid covers exactly nrows/128 blocks; any remainder goes to the tail kernel.
__global__ __launch_bounds__(256) void MRBF_rbf_wmma_kernel(
    const float* __restrict__ x,
    const float* __restrict__ centres,
    const float* __restrict__ log_sigmas,
    float* __restrict__ out)
{
    __shared__ _Float16 cenH[OUT_F * ROW_H];   // -2 * centres, f16, padded rows
    __shared__ float    c2iv[OUT_F * 2];       // interleaved {||c||^2, inv_var * log2(e)}
    __shared__ float    x2s[8 * 32];           // per-wave ||x||^2 partials

    const int tid = threadIdx.x;

    // ---- Stage centres -> LDS (f16, scaled by -2), plus c2 and inv_var*log2e ----
    if (tid < OUT_F) {
        const float* crow = centres + tid * IN_F;
        float c2 = 0.0f;
        #pragma unroll
        for (int k = 0; k < IN_F; k += 4) {
            float4 v = *(const float4*)(crow + k);
            c2 = fmaf(v.x, v.x, c2);
            c2 = fmaf(v.y, v.y, c2);
            c2 = fmaf(v.z, v.z, c2);
            c2 = fmaf(v.w, v.w, c2);
            cenH[tid * ROW_H + k + 0] = (_Float16)(-2.0f * v.x);
            cenH[tid * ROW_H + k + 1] = (_Float16)(-2.0f * v.y);
            cenH[tid * ROW_H + k + 2] = (_Float16)(-2.0f * v.z);
            cenH[tid * ROW_H + k + 3] = (_Float16)(-2.0f * v.w);
        }
        c2iv[2 * tid + 0] = c2;
        float iv = __builtin_amdgcn_exp2f(-2.0f * log_sigmas[tid] * LOG2E); // exp(-2 log_sigma)
        c2iv[2 * tid + 1] = iv * LOG2E;
    }
    __syncthreads();

    const int lane = tid & 31;
    const int wave = tid >> 5;
    const int hi   = lane >> 4;        // 0: K 0-7 & 16-23, 1: K 8-15 & 24-31
    const int lm   = lane & 15;
    const int rowBase = blockIdx.x * 128 + wave * 16;

    // ---- A fragment: 16x32 f16 (rows rowBase..rowBase+15) ----
    // 16-bit A layout: lanes 0-15 hold K=0..7 (v0-3) and K=16..23 (v4-7);
    // lanes 16-31 hold K=8..15 and K=24..31; row M = lane & 15.
    const int kb = hi * 8;
    const float* xr = x + (size_t)(rowBase + lm) * IN_F;
    float4 f0 = *(const float4*)(xr + kb + 0);
    float4 f1 = *(const float4*)(xr + kb + 4);
    float4 f2 = *(const float4*)(xr + kb + 16);
    float4 f3 = *(const float4*)(xr + kb + 20);

    v16h a;
    a[0]  = (_Float16)f0.x; a[1]  = (_Float16)f0.y; a[2]  = (_Float16)f0.z; a[3]  = (_Float16)f0.w;
    a[4]  = (_Float16)f1.x; a[5]  = (_Float16)f1.y; a[6]  = (_Float16)f1.z; a[7]  = (_Float16)f1.w;
    a[8]  = (_Float16)f2.x; a[9]  = (_Float16)f2.y; a[10] = (_Float16)f2.z; a[11] = (_Float16)f2.w;
    a[12] = (_Float16)f3.x; a[13] = (_Float16)f3.y; a[14] = (_Float16)f3.z; a[15] = (_Float16)f3.w;

    // ---- ||x||^2: lane j holds half of row (j&15); combine via per-wave LDS slab ----
    float part = 0.0f;
    part = fmaf(f0.x, f0.x, part); part = fmaf(f0.y, f0.y, part);
    part = fmaf(f0.z, f0.z, part); part = fmaf(f0.w, f0.w, part);
    part = fmaf(f1.x, f1.x, part); part = fmaf(f1.y, f1.y, part);
    part = fmaf(f1.z, f1.z, part); part = fmaf(f1.w, f1.w, part);
    part = fmaf(f2.x, f2.x, part); part = fmaf(f2.y, f2.y, part);
    part = fmaf(f2.z, f2.z, part); part = fmaf(f2.w, f2.w, part);
    part = fmaf(f3.x, f3.x, part); part = fmaf(f3.y, f3.y, part);
    part = fmaf(f3.z, f3.z, part); part = fmaf(f3.w, f3.w, part);

    float* ws = x2s + wave * 32;
    ws[lane] = part;                      // same-wave LDS ops are in-order
    // x2[8*hi + r] = part[8*hi + r] + part[8*hi + r + 16]; broadcast reads, conflict-free.
    float4 pl0 = *(const float4*)(ws + hi * 8 + 0);
    float4 pl1 = *(const float4*)(ws + hi * 8 + 4);
    float4 ph0 = *(const float4*)(ws + 16 + hi * 8 + 0);
    float4 ph1 = *(const float4*)(ws + 16 + hi * 8 + 4);
    float x2bc[8];
    x2bc[0] = pl0.x + ph0.x; x2bc[1] = pl0.y + ph0.y;
    x2bc[2] = pl0.z + ph0.z; x2bc[3] = pl0.w + ph0.w;
    x2bc[4] = pl1.x + ph1.x; x2bc[5] = pl1.y + ph1.y;
    x2bc[6] = pl1.z + ph1.z; x2bc[7] = pl1.w + ph1.w;

    float* orow = out + (size_t)(rowBase + hi * 8) * OUT_F + lm;

    #pragma unroll
    for (int t = 0; t < 8; ++t) {
        const int n = t * 16 + lm;                 // output column for this lane
        // ---- B fragment from LDS: 32x16 f16 tile (pre-scaled -2*c) ----
        const _Float16* bp = cenH + n * ROW_H + hi * 8;
        v8h blo = *(const v8h*)(bp);
        v8h bhi = *(const v8h*)(bp + 16);
        v16h b = __builtin_shufflevector(blo, bhi,
                                         0, 1, 2, 3, 4, 5, 6, 7,
                                         8, 9, 10, 11, 12, 13, 14, 15);

        float2 ci = *(const float2*)(c2iv + 2 * n); // {c2, inv_var*log2e}

        // C init: x^2[row] + c^2[col]  ->  D = sqdist
        v8f acc;
        #pragma unroll
        for (int r = 0; r < 8; ++r) acc[r] = x2bc[r] + ci.x;

        acc = __builtin_amdgcn_wmma_f32_16x16x32_f16(
            /*neg_a=*/false, a, /*neg_b=*/false, b,
            /*c_mod=*/(short)0, acc, /*reuse_a=*/false, /*reuse_b=*/false);

        #pragma unroll
        for (int r = 0; r < 8; ++r) {
            float s   = __builtin_amdgcn_fmed3f(acc[r], 0.0f, FLT_MAX); // clamp >= 0, 1 op
            float val = __builtin_amdgcn_exp2f(-s * ci.y);
            orow[(size_t)r * OUT_F + t * 16] = val;
        }
    }
}

// Scalar tail for nrows % 128 != 0 (never taken for N = 1048576).
__global__ void MRBF_rbf_tail_kernel(
    const float* __restrict__ x,
    const float* __restrict__ centres,
    const float* __restrict__ log_sigmas,
    float* __restrict__ out,
    int rowStart, int nrows)
{
    int idx   = blockIdx.x * blockDim.x + threadIdx.x;
    int total = (nrows - rowStart) * OUT_F;
    if (idx >= total) return;
    int row = rowStart + idx / OUT_F;
    int col = idx - (idx / OUT_F) * OUT_F;
    const float* xr = x + (size_t)row * IN_F;
    const float* cr = centres + (size_t)col * IN_F;
    float s = 0.0f;
    #pragma unroll
    for (int k = 0; k < IN_F; ++k) {
        float d = xr[k] - cr[k];
        s = fmaf(d, d, s);
    }
    float iv = __builtin_amdgcn_exp2f(-2.0f * log_sigmas[col] * LOG2E);
    out[(size_t)row * OUT_F + col] = __builtin_amdgcn_exp2f(-s * iv * LOG2E);
}

extern "C" void kernel_launch(void* const* d_in, const int* in_sizes, int n_in,
                              void* d_out, int out_size, void* d_ws, size_t ws_size,
                              hipStream_t stream) {
    const float* x          = (const float*)d_in[0];
    const float* centres    = (const float*)d_in[1];
    const float* log_sigmas = (const float*)d_in[2];
    float* out = (float*)d_out;

    const int nrows    = in_sizes[0] / IN_F;   // N (1048576)
    const int mainBlks = nrows / 128;          // 8192 full blocks, no guards
    const int mainRows = mainBlks * 128;
    const int remRows  = nrows - mainRows;

    if (mainBlks > 0)
        MRBF_rbf_wmma_kernel<<<mainBlks, 256, 0, stream>>>(x, centres, log_sigmas, out);
    if (remRows > 0) {
        int tailThreads = remRows * OUT_F;
        int tailBlks    = (tailThreads + 255) / 256;
        MRBF_rbf_tail_kernel<<<tailBlks, 256, 0, stream>>>(x, centres, log_sigmas, out,
                                                           mainRows, nrows);
    }
}
